// MultiheadCrossAttention_67534065762564
// MI455X (gfx1250) — matrix operand order repaired and checked by reading, not compiled, not run on hardware
//
#include <hip/hip_runtime.h>
#include <hip/hip_bf16.h>

typedef __attribute__((ext_vector_type(16))) _Float16 v16h;
typedef __attribute__((ext_vector_type(8)))  float    v8f;

#define D_MODEL 1024
#define NHEADS  16
#define HDIM    64
#define BATCH   4
#define SEQ     1024

// Fragment unions. For 16-bit A-fragments the per-lane data is two contiguous
// 8-half (16B) runs: halves [8*grp, 8*grp+8) and [16+8*grp, 16+8*grp+8).
// B-fragments use the same two 8-half runs (k order must match A).
union AFrag { v16h v; unsigned u[8]; uint4 q[2]; _Float16 h[16]; };
union CFrag { v8f  v; float    f[8]; };

__device__ __forceinline__ float half16_max(float x) {
#pragma unroll
  for (int off = 1; off < 16; off <<= 1) x = fmaxf(x, __shfl_xor(x, off, 32));
  return x;
}
__device__ __forceinline__ float half16_sum(float x) {
#pragma unroll
  for (int off = 1; off < 16; off <<= 1) x += __shfl_xor(x, off, 32);
  return x;
}

// Load A-fragment (16x32 f16) from a 16B-aligned row of halves.
__device__ __forceinline__ void load_afrag(AFrag& a, const _Float16* row,
                                           int grp) {
  a.q[0] = *(const uint4*)(row + 8 * grp);
  a.q[1] = *(const uint4*)(row + 16 + 8 * grp);
}
// Load B-fragment (32x16 f16) from a 16B-aligned row of halves ([n][k] layout).
__device__ __forceinline__ void load_bfrag(AFrag& b, const _Float16* row,
                                           int grp) {
  b.q[0] = *(const uint4*)(row + 8 * grp);
  b.q[1] = *(const uint4*)(row + 16 + 8 * grp);
}

// C[M,N] = A[M,K] @ W[K,N] + bias, double-buffered LDS, one barrier per k-step.
// MODE 0: out f16 Q layout [b][h][s][hd], value=(acc+bias)*0.125 (scale folded)
// MODE 1: out f16 K/V split from fused KV gemm, layouts [b][h][s][hd]
// MODE 2: out fp32 row-major [m][n]
template <int MODE, typename AT>
__global__ __launch_bounds__(128) void gemm_wmma_kernel(
    const AT* __restrict__ A, const float* __restrict__ W,
    const float* __restrict__ bias,
    _Float16* __restrict__ outH0, _Float16* __restrict__ outH1,
    float* __restrict__ outF, int M, int N, int K) {
  __shared__ __align__(16) _Float16 As[2][64][72];  // [m][k], 144B rows
  __shared__ __align__(16) _Float16 Bs[2][64][72];  // [n][k] (W transposed)

  const int tid  = threadIdx.x;
  const int wv   = tid >> 5;
  const int lane = tid & 31;
  const int l16  = lane & 15;
  const int grp  = lane >> 4;

  const int m0 = blockIdx.x * 64;
  const int n0 = blockIdx.y * 64;

  CFrag acc[4];
#pragma unroll
  for (int nt = 0; nt < 4; ++nt)
#pragma unroll
    for (int r = 0; r < 8; ++r) acc[nt].f[r] = 0.f;

  auto stage = [&](int db, int k0) {
#pragma unroll
    for (int i = 0; i < 16; ++i) {
      int e = i * 128 + tid;
      int r = e >> 5, c = e & 31;
      As[db][r][c] = (_Float16)A[(size_t)(m0 + r) * K + (k0 + c)];
    }
#pragma unroll
    for (int i = 0; i < 16; ++i) {
      int e = i * 128 + tid;
      int kr = e >> 6, nc = e & 63;
      Bs[db][nc][kr] = (_Float16)W[(size_t)(k0 + kr) * N + (n0 + nc)];
    }
  };

  stage(0, 0);
  const int nsteps = K >> 5;
  for (int kk = 0; kk < nsteps; ++kk) {
    __syncthreads();
    const int db = kk & 1;
    if (kk + 1 < nsteps) {
      stage(db ^ 1, (kk + 1) * 32);
      if (kk + 2 < nsteps) {  // gfx1250 global_prefetch for tile after next
        const int kp = (kk + 2) * 32;
        __builtin_prefetch(&A[(size_t)(m0 + (tid >> 1)) * K + kp], 0, 0);
        __builtin_prefetch(&W[(size_t)(kp + (tid & 31)) * N + n0], 0, 0);
      }
    }

    AFrag a;
    load_afrag(a, &As[db][wv * 16 + l16][0], grp);

#pragma unroll
    for (int nt = 0; nt < 4; ++nt) {
      AFrag bfr;
      load_bfrag(bfr, &Bs[db][nt * 16 + l16][0], grp);
      acc[nt].v = __builtin_amdgcn_wmma_f32_16x16x32_f16(
          false, a.v, false, bfr.v, (short)0, acc[nt].v, false, false);
    }
  }

  // epilogue: C layout row = r + 8*grp, col = l16
#pragma unroll
  for (int nt = 0; nt < 4; ++nt) {
    const int n = n0 + nt * 16 + l16;
    const float bv = bias[n];
#pragma unroll
    for (int r = 0; r < 8; ++r) {
      const int m = m0 + wv * 16 + r + 8 * grp;
      float val = acc[nt].f[r] + bv;
      if constexpr (MODE == 0) {
        int bb = m >> 10, s = m & 1023;
        int h = n >> 6, d = n & 63;
        outH0[((size_t)(bb * NHEADS + h) * SEQ + s) * HDIM + d] =
            (_Float16)(val * 0.125f);
      } else if constexpr (MODE == 1) {
        int bb = m >> 10, s = m & 1023;
        int h = n >> 7, dd = n & 127;
        _Float16* dst = (dd < 64) ? outH0 : outH1;
        dst[((size_t)(bb * NHEADS + h) * SEQ + s) * HDIM + (dd & 63)] =
            (_Float16)val;
      } else {
        outF[(size_t)m * N + n] = val;
      }
    }
  }
}

// Flash attention: grid (S/64, B*H), 4 waves each own 16 q rows; 32-key chunks,
// double-buffered: K staged via GLOBAL_LOAD_ASYNC_TO_LDS_B128 (ASYNCcnt),
// V staged transposed with scalar ds stores.
__global__ __launch_bounds__(128) void attn_wmma_kernel(
    const _Float16* __restrict__ q16, const _Float16* __restrict__ k16,
    const _Float16* __restrict__ v16, const float* __restrict__ mask,
    _Float16* __restrict__ o16) {
  __shared__ __align__(16) _Float16 Ks[2][32][72];  // [key j][d], 144B rows
  __shared__ __align__(16) _Float16 Vs[2][64][72];  // [d][key j] (transposed)
  __shared__ __align__(16) _Float16 Ps[4][16][72];  // per-wave P tile [m][j]

  const int tid  = threadIdx.x;
  const int wv   = tid >> 5;
  const int lane = tid & 31;
  const int l16  = lane & 15;
  const int grp  = lane >> 4;

  const int bh = blockIdx.y;
  const int bb = bh >> 4, h = bh & 15;
  const int m0 = blockIdx.x * 64;

  const _Float16* Q  = q16 + (size_t)bh * SEQ * HDIM;
  const _Float16* Kp = k16 + (size_t)bh * SEQ * HDIM;
  const _Float16* Vp = v16 + (size_t)bh * SEQ * HDIM;

  const int kj   = tid >> 2;        // 0..31: key row this thread stages
  const int kseg = (tid & 3) * 16;  // 16-half (32B) segment within the row

  auto issueK = [&](int db, int c0) {
    unsigned ldsa = (unsigned)(size_t)&Ks[db][kj][kseg];
    const _Float16* g = Kp + (size_t)(c0 + kj) * HDIM + kseg;
    asm volatile("global_load_async_to_lds_b128 %0, %1, off" ::"v"(ldsa),
                 "v"(g)
                 : "memory");
    asm volatile("global_load_async_to_lds_b128 %0, %1, off" ::"v"(ldsa + 16u),
                 "v"(g + 8)
                 : "memory");
  };
  auto stageV = [&](int db, int c0) {
#pragma unroll
    for (int i = 0; i < 16; ++i) {
      int e = i * 128 + tid;
      int j = e >> 6, d = e & 63;
      Vs[db][d][j] = Vp[(size_t)(c0 + j) * HDIM + d];
    }
  };

  // Q fragments: 16 rows x 64 -> two A frags (K=32 each), loaded once (b128s).
  const int srow = m0 + wv * 16 + l16;
  AFrag qa[2];
#pragma unroll
  for (int f = 0; f < 2; ++f)
    load_afrag(qa[f], Q + (size_t)srow * HDIM + 32 * f, grp);

  float mrow[8], lrow[8];
  CFrag o[4];
#pragma unroll
  for (int r = 0; r < 8; ++r) { mrow[r] = -1e30f; lrow[r] = 0.f; }
#pragma unroll
  for (int nt = 0; nt < 4; ++nt)
#pragma unroll
    for (int r = 0; r < 8; ++r) o[nt].f[r] = 0.f;

  issueK(0, 0);
  stageV(0, 0);

  for (int c = 0; c < SEQ / 32; ++c) {
    const int c0 = c * 32;
    const int db = c & 1;
    // drain this chunk's async K loads, then make all staging visible
    asm volatile("s_wait_asynccnt 0x0" ::: "memory");
    __syncthreads();
    // pipeline next chunk (buffer was fully consumed before the barrier above)
    if (c0 + 32 < SEQ) {
      issueK(db ^ 1, c0 + 32);
      stageV(db ^ 1, c0 + 32);
    }

    // logits: two 16-key n-tiles, chained over d (K=64 -> 2 WMMAs each)
    CFrag sc[2];
#pragma unroll
    for (int jt = 0; jt < 2; ++jt) {
#pragma unroll
      for (int r = 0; r < 8; ++r) sc[jt].f[r] = 0.f;
#pragma unroll
      for (int ks = 0; ks < 2; ++ks) {
        AFrag kb;
        load_bfrag(kb, &Ks[db][jt * 16 + l16][32 * ks], grp);
        sc[jt].v = __builtin_amdgcn_wmma_f32_16x16x32_f16(
            false, qa[ks].v, false, kb.v, (short)0, sc[jt].v, false, false);
      }
      const int kcol = c0 + jt * 16 + l16;
#pragma unroll
      for (int r = 0; r < 8; ++r) {
        const int sq = m0 + wv * 16 + r + 8 * grp;
        sc[jt].f[r] += mask[(size_t)sq * SEQ + kcol];
      }
    }

    // online softmax; fold both tiles per-lane before the 16-lane butterflies
#pragma unroll
    for (int r = 0; r < 8; ++r) {
      float tmax = half16_max(fmaxf(sc[0].f[r], sc[1].f[r]));
      float mnew = fmaxf(mrow[r], tmax);
      float scale = __expf(mrow[r] - mnew);
      mrow[r] = mnew;
      float p0 = __expf(sc[0].f[r] - mnew);
      float p1 = __expf(sc[1].f[r] - mnew);
      lrow[r] = lrow[r] * scale + half16_sum(p0 + p1);
#pragma unroll
      for (int nt = 0; nt < 4; ++nt) o[nt].f[r] *= scale;
      const int pm = r + 8 * grp;
      Ps[wv][pm][l16]      = (_Float16)p0;
      Ps[wv][pm][16 + l16] = (_Float16)p1;
    }

    // PV: A = P(16x32) from per-wave LDS, B = V(32 x 64) from transposed LDS
    AFrag pa;
    load_afrag(pa, &Ps[wv][l16][0], grp);
#pragma unroll
    for (int nt = 0; nt < 4; ++nt) {
      AFrag vb;
      load_bfrag(vb, &Vs[db][nt * 16 + l16][0], grp);
      o[nt].v = __builtin_amdgcn_wmma_f32_16x16x32_f16(
          false, pa.v, false, vb.v, (short)0, o[nt].v, false, false);
    }
  }

  // finalize: divide by row sum, store f16 [b][s][h*64+d]
#pragma unroll
  for (int r = 0; r < 8; ++r) {
    const float inv = 1.f / lrow[r];
    const int sq = m0 + wv * 16 + r + 8 * grp;
#pragma unroll
    for (int nt = 0; nt < 4; ++nt) {
      const int d = nt * 16 + l16;
      o16[(size_t)(bb * SEQ + sq) * D_MODEL + h * HDIM + d] =
          (_Float16)(o[nt].f[r] * inv);
    }
  }
}

extern "C" void kernel_launch(void* const* d_in, const int* in_sizes, int n_in,
                              void* d_out, int out_size, void* d_ws,
                              size_t ws_size, hipStream_t stream) {
  const float* x     = (const float*)d_in[0];
  const float* y     = (const float*)d_in[1];
  const float* mask  = (const float*)d_in[2];
  const float* q_w   = (const float*)d_in[3];
  const float* q_b   = (const float*)d_in[4];
  const float* kv_w  = (const float*)d_in[5];
  const float* kv_b  = (const float*)d_in[6];
  const float* out_w = (const float*)d_in[7];
  const float* out_b = (const float*)d_in[8];

  const size_t elems = (size_t)BATCH * SEQ * D_MODEL;  // 4M halves each
  _Float16* q16 = (_Float16*)d_ws;
  _Float16* k16 = q16 + elems;
  _Float16* v16 = k16 + elems;
  _Float16* a16 = v16 + elems;

  // Q = x @ q_w + q_b  (scale folded), head-major f16
  gemm_wmma_kernel<0, float><<<dim3(64, 16), 128, 0, stream>>>(
      x, q_w, q_b, q16, nullptr, nullptr, BATCH * SEQ, D_MODEL, D_MODEL);
  // KV = y @ kv_w + kv_b, split into K / V head-major f16
  gemm_wmma_kernel<1, float><<<dim3(64, 32), 128, 0, stream>>>(
      y, kv_w, kv_b, k16, v16, nullptr, BATCH * SEQ, 2 * D_MODEL, D_MODEL);
  // flash attention -> a16 [b][s][h*hd] f16
  attn_wmma_kernel<<<dim3(SEQ / 64, BATCH * NHEADS), 128, 0, stream>>>(
      q16, k16, v16, mask, a16);
  // out = a16 @ out_w + out_b (fp32)
  gemm_wmma_kernel<2, _Float16><<<dim3(64, 16), 128, 0, stream>>>(
      a16, out_w, out_b, nullptr, nullptr, (float*)d_out, BATCH * SEQ, D_MODEL,
      D_MODEL);
}
